// RGCNLayer_30829275250940
// MI455X (gfx1250) — matrix operand order repaired and checked by reading, not compiled, not run on hardware
//
#include <hip/hip_runtime.h>
#include <hip/hip_bf16.h>

// ---------------------------------------------------------------------------
// RGCN layer for MI455X (gfx1250, wave32, WMMA).
//
// h[dst] += (x[src]*norm) @ W[etype]   via relation-binned bf16 WMMA tiles
// y = relu(h+bias) + relu(x@res_w^T+res_b); out = batchnorm(y)
// ---------------------------------------------------------------------------

typedef __attribute__((ext_vector_type(16))) __bf16 v16bf;
typedef __attribute__((ext_vector_type(8)))  __bf16 v8bf;
typedef __attribute__((ext_vector_type(8)))  float  v8f;

#define F 64  // IN_F == OUT_F == 64

__device__ __forceinline__ v8f wmma_bf16(v16bf a, v16bf b, v8f c) {
  // D = A(16x32) * B(32x16) + C, f32 accumulate
  return __builtin_amdgcn_wmma_f32_16x16x32_bf16(
      /*neg_a=*/false, a, /*neg_b=*/false, b,
      /*c_mod=*/(short)0, c, /*reuse_a=*/false, /*reuse_b=*/false);
}

// A-fragment (16x32 bf16). Lane m (=lane&15) holds row m. For the half-wave
// hi = lane>>4:  slots 0..15 -> K = kbase + 16*hi + s.  Contiguous per lane.
__device__ __forceinline__ v16bf make_a_frag(const float* __restrict__ row,
                                             float scale, int hi, int kbase) {
  const float4* p = reinterpret_cast<const float4*>(row + kbase + 16 * hi);
  const float4* q = reinterpret_cast<const float4*>(row + kbase + 16 * hi + 8);
  float4 a = p[0], b = p[1], c = q[0], d = q[1];
  float f[16] = {a.x, a.y, a.z, a.w, b.x, b.y, b.z, b.w,
                 c.x, c.y, c.z, c.w, d.x, d.y, d.z, d.w};
  v16bf r;
#pragma unroll
  for (int s = 0; s < 16; ++s) r[s] = (__bf16)(f[s] * scale);
  return r;
}

// B-fragment (32x16 bf16) from a bf16 column stored contiguously in K:
// lane (n = lane&15, hi) holds column n, elements K = 16*hi + s (s=0..15).
// Caller passes p = &Wcol[16*hi (+32*kb)] -> 16 contiguous bf16 = 32B.
__device__ __forceinline__ v16bf load_b_frag(const __bf16* __restrict__ p) {
  v8bf lo = *reinterpret_cast<const v8bf*>(p);
  v8bf hi = *reinterpret_cast<const v8bf*>(p + 8);
  return __builtin_shufflevector(lo, hi, 0, 1, 2, 3, 4, 5, 6, 7, 8, 9, 10, 11,
                                 12, 13, 14, 15);
}

// Same B-fragment but converting from 16 contiguous f32.
__device__ __forceinline__ v16bf make_b_frag_f32(const float* __restrict__ p) {
  const float4* q = reinterpret_cast<const float4*>(p);
  float4 a = q[0], b = q[1], c = q[2], d = q[3];
  float f[16] = {a.x, a.y, a.z, a.w, b.x, b.y, b.z, b.w,
                 c.x, c.y, c.z, c.w, d.x, d.y, d.z, d.w};
  v16bf r;
#pragma unroll
  for (int s = 0; s < 16; ++s) r[s] = (__bf16)f[s];
  return r;
}

// ---------------------------------------------------------------------------
// Phase 0: zero accumulators, fill permutation with -1
__global__ void k_init(float* __restrict__ h, int hN, int* __restrict__ perm,
                       int permN, int* __restrict__ counts,
                       float* __restrict__ sums, float* __restrict__ sumsq,
                       int R) {
  int idx = blockIdx.x * blockDim.x + threadIdx.x;
  if (idx < hN) h[idx] = 0.0f;
  if (idx < permN) perm[idx] = -1;
  if (idx < R) { counts[idx] = 0; sums[idx] = 0.0f; sumsq[idx] = 0.0f; }
}

// Phase 1: compose W_r = sum_b comp[r,b] * basis[b]; store bf16 TRANSPOSED
// as Wbf[r][o][i] so a WMMA B-column (fixed o, K contiguous) is contiguous.
__global__ void k_compose(const float* __restrict__ comp,
                          const float* __restrict__ basis,
                          __bf16* __restrict__ Wbf, int B) {
  int r = blockIdx.x;
  __shared__ float cs[256];
  for (int b = threadIdx.x; b < B; b += blockDim.x) cs[b] = comp[r * B + b];
  __syncthreads();
  for (int idx = threadIdx.x; idx < F * F; idx += blockDim.x) {
    int i = idx >> 6, o = idx & 63;
    float acc = 0.0f;
    for (int b = 0; b < B; ++b) acc += cs[b] * basis[(size_t)b * F * F + idx];
    Wbf[(size_t)r * F * F + o * F + i] = (__bf16)acc;
  }
}

// Phase 2: per-relation histogram
__global__ void k_hist(const int* __restrict__ etype, int* __restrict__ counts,
                       int E) {
  int e = blockIdx.x * blockDim.x + threadIdx.x;
  if (e < E) atomicAdd(&counts[etype[e]], 1);
}

// Phase 3: 16-aligned exclusive scan over 64 bins (trivial serial)
__global__ void k_scan(const int* __restrict__ counts, int* __restrict__ pad_off,
                       int* __restrict__ cursors, int R) {
  if (threadIdx.x == 0) {
    int acc = 0;
    for (int r = 0; r < R; ++r) {
      pad_off[r] = acc;
      cursors[r] = acc;
      acc += (counts[r] + 15) & ~15;
    }
    pad_off[R] = acc;
  }
}

// Phase 4: scatter edge ids into relation bins
__global__ void k_scatter(const int* __restrict__ etype,
                          int* __restrict__ cursors, int* __restrict__ perm,
                          int E) {
  int e = blockIdx.x * blockDim.x + threadIdx.x;
  if (e < E) {
    int pos = atomicAdd(&cursors[etype[e]], 1);
    perm[pos] = e;
  }
}

// Phase 5: the hot loop. One wave per 16-edge tile (all edges in a tile share
// one relation by construction). A = 16 gathered source rows * norm (bf16),
// B = W_rel column tiles (all 8 fragments preloaded so the loads clause and
// overlap the A conversion), D scattered with UNGUARDED global f32 atomics:
// padded rows have norm==0 so their entire D row is exactly 0.0 and the
// atomicAdd to h[0][*] is a numeric no-op. This keeps EXEC all-ones end to
// end (no saveexec/branch per element).
__global__ void __launch_bounds__(256) k_edge_gemm(
    const float* __restrict__ nf, const int* __restrict__ src,
    const int* __restrict__ dst, const int* __restrict__ etype,
    const float* __restrict__ norm, const __bf16* __restrict__ Wbf,
    const int* __restrict__ perm, const int* __restrict__ pad_off,
    float* __restrict__ h, int R) {
  int wave = blockIdx.x * (blockDim.x >> 5) + (threadIdx.x >> 5);
  int lane = threadIdx.x & 31;
  int total_pad = pad_off[R];     // wave-uniform
  int base = wave * 16;
  if (base >= total_pad) return;  // uniform exit: EXEC stays all-ones for WMMA
  int m = lane & 15, hi = lane >> 4;

  int rel = etype[perm[base]];    // first slot of every tile is a valid edge

  // This lane's A-row (row m of the tile); padded slots contribute zero.
  int em = perm[base + m];
  int srow = 0;
  float nrm = 0.0f;
  if (em >= 0) { srow = src[em]; nrm = norm[em]; }

  // Output rows handled by this lane: M = v + 8*hi. Padded -> dst 0 (adds 0).
  int drow[8];
#pragma unroll
  for (int v = 0; v < 8; ++v) {
    int ee = perm[base + v + 8 * hi];
    drow[v] = (ee >= 0) ? dst[ee] : 0;
  }

  __builtin_prefetch(perm + base + 128, 0, 0);  // speculative, next tiles

  // Preload all 8 B fragments (4 n-tiles x 2 k-steps) so loads clause.
  const __bf16* wrel = Wbf + (size_t)rel * F * F;
  v16bf Bf[4][2];
#pragma unroll
  for (int nb = 0; nb < 4; ++nb) {
    const __bf16* col = wrel + (size_t)(nb * 16 + m) * F + 16 * hi;
    Bf[nb][0] = load_b_frag(col);       // K 0..31 of this column
    Bf[nb][1] = load_b_frag(col + 32);  // K 32..63
  }

  const float* row = nf + (size_t)srow * F;
  v16bf A0 = make_a_frag(row, nrm, hi, 0);   // K 0..31
  v16bf A1 = make_a_frag(row, nrm, hi, 32);  // K 32..63

#pragma unroll
  for (int nb = 0; nb < 4; ++nb) {
    v8f acc = {};
    acc = wmma_bf16(A0, Bf[nb][0], acc);
    acc = wmma_bf16(A1, Bf[nb][1], acc);
#pragma unroll
    for (int v = 0; v < 8; ++v)
      atomicAdd(&h[(size_t)drow[v] * F + nb * 16 + m], acc[v]);
  }
}

// Phase 6: residual GEMM (WMMA) fused with bias + ReLU + add and the
// batch-norm column sum/sumsq reduction. Writes pre-BN y into d_out.
__global__ void __launch_bounds__(256) k_residual(
    const float* __restrict__ nf, const float* __restrict__ h,
    const float* __restrict__ bias, const float* __restrict__ res_w,
    const float* __restrict__ res_b, float* __restrict__ y,
    float* __restrict__ sums, float* __restrict__ sumsq, int N) {
  int wave = blockIdx.x * (blockDim.x >> 5) + (threadIdx.x >> 5);
  int lane = threadIdx.x & 31;
  int tiles = (N + 15) >> 4;
  if (wave >= tiles) return;  // wave-uniform
  int m = lane & 15, hi = lane >> 4;
  int base = wave * 16;
  bool full = (base + 16 <= N);  // wave-uniform fast path (N%16==0 -> always)

  int rowm = base + m;
  if (rowm >= N) rowm = N - 1;  // duplicate row; stores masked below
  const float* arow = nf + (size_t)rowm * F;
  v16bf A0 = make_a_frag(arow, 1.0f, hi, 0);
  v16bf A1 = make_a_frag(arow, 1.0f, hi, 32);

  // Preload all B fragments (res_w is [out][in] row-major: column o of the
  // WMMA B matrix is res_w row o, K contiguous).
  v16bf Bf[4][2];
#pragma unroll
  for (int nb = 0; nb < 4; ++nb) {
    const float* wr = res_w + (size_t)(nb * 16 + m) * F + 16 * hi;
    Bf[nb][0] = make_b_frag_f32(wr);
    Bf[nb][1] = make_b_frag_f32(wr + 32);
  }

#pragma unroll
  for (int nb = 0; nb < 4; ++nb) {
    int col = nb * 16 + m;
    v8f acc = {};
    acc = wmma_bf16(A0, Bf[nb][0], acc);
    acc = wmma_bf16(A1, Bf[nb][1], acc);

    float bcol = bias[col], rbcol = res_b[col];
    float s = 0.0f, s2 = 0.0f;
    if (full) {
#pragma unroll
      for (int v = 0; v < 8; ++v) {
        int r = base + v + 8 * hi;
        float hv = h[(size_t)r * F + col];
        float yv = fmaxf(hv + bcol, 0.0f) + fmaxf(acc[v] + rbcol, 0.0f);
        y[(size_t)r * F + col] = yv;
        s += yv;
        s2 += yv * yv;
      }
    } else {
#pragma unroll
      for (int v = 0; v < 8; ++v) {
        int r = base + v + 8 * hi;
        if (r < N) {
          float hv = h[(size_t)r * F + col];
          float yv = fmaxf(hv + bcol, 0.0f) + fmaxf(acc[v] + rbcol, 0.0f);
          y[(size_t)r * F + col] = yv;
          s += yv;
          s2 += yv * yv;
        }
      }
    }
    atomicAdd(&sums[col], s);
    atomicAdd(&sumsq[col], s2);
  }
}

// Phase 7: batch-norm in place on d_out
__global__ void k_bn(float* __restrict__ y, const float* __restrict__ sums,
                     const float* __restrict__ sumsq,
                     const float* __restrict__ gamma,
                     const float* __restrict__ beta, int N) {
  int idx = blockIdx.x * blockDim.x + threadIdx.x;
  int total = N * F;
  if (idx >= total) return;
  int c = idx & 63;
  float invN = 1.0f / (float)N;
  float mean = sums[c] * invN;
  float var = sumsq[c] * invN - mean * mean;
  y[idx] = (y[idx] - mean) * rsqrtf(var + 1e-5f) * gamma[c] + beta[c];
}

// ---------------------------------------------------------------------------
extern "C" void kernel_launch(void* const* d_in, const int* in_sizes, int n_in,
                              void* d_out, int out_size, void* d_ws,
                              size_t ws_size, hipStream_t stream) {
  const float* nf    = (const float*)d_in[0];
  const int*   src   = (const int*)d_in[1];
  const int*   dst   = (const int*)d_in[2];
  const int*   etype = (const int*)d_in[3];
  const float* norm  = (const float*)d_in[4];
  const float* basis = (const float*)d_in[5];
  const float* comp  = (const float*)d_in[6];
  const float* bias  = (const float*)d_in[7];
  const float* res_w = (const float*)d_in[8];
  const float* res_b = (const float*)d_in[9];
  const float* gamma = (const float*)d_in[10];
  const float* beta  = (const float*)d_in[11];

  const int N = in_sizes[0] / F;
  const int E = in_sizes[1];
  const int B = in_sizes[5] / (F * F);
  const int R = in_sizes[6] / B;

  // Workspace carve-up (~20 MB for the reference sizes)
  char* ws = (char*)d_ws;
  size_t off = 0;
  int*   counts  = (int*)(ws + off);  off += 256;
  int*   pad_off = (int*)(ws + off);  off += 512;   // R+1 ints
  int*   cursors = (int*)(ws + off);  off += 256;
  float* sums    = (float*)(ws + off); off += 256;
  float* sumsq   = (float*)(ws + off); off += 256;
  off = (off + 255) & ~(size_t)255;
  __bf16* Wbf = (__bf16*)(ws + off);  off += (size_t)R * F * F * 2;
  off = (off + 255) & ~(size_t)255;
  int permN = E + R * 16;
  int* perm = (int*)(ws + off);       off += (size_t)permN * 4;
  off = (off + 255) & ~(size_t)255;
  float* h = (float*)(ws + off);      off += (size_t)N * F * 4;

  float* y = (float*)d_out;

  int hN = N * F;
  int initN = hN > permN ? hN : permN;
  k_init<<<(initN + 255) / 256, 256, 0, stream>>>(h, hN, perm, permN, counts,
                                                  sums, sumsq, R);
  k_compose<<<R, 256, 0, stream>>>(comp, basis, Wbf, B);
  k_hist<<<(E + 255) / 256, 256, 0, stream>>>(etype, counts, E);
  k_scan<<<1, 64, 0, stream>>>(counts, pad_off, cursors, R);
  k_scatter<<<(E + 255) / 256, 256, 0, stream>>>(etype, cursors, perm, E);

  int maxTiles = (E + 15) / 16 + R;  // upper bound on padded tiles
  k_edge_gemm<<<(maxTiles + 7) / 8, 256, 0, stream>>>(nf, src, dst, etype, norm,
                                                      Wbf, perm, pad_off, h, R);
  int ntiles = (N + 15) / 16;
  k_residual<<<(ntiles + 7) / 8, 256, 0, stream>>>(nf, h, bias, res_w, res_b, y,
                                                   sums, sumsq, N);
  k_bn<<<(hN + 255) / 256, 256, 0, stream>>>(y, sums, sumsq, gamma, beta, N);
}
